// FlashMMSequenceMixing_52785148068656
// MI455X (gfx1250) — matrix operand order repaired and checked by reading, not compiled, not run on hardware
//
#include <hip/hip_runtime.h>
#include <hip/hip_bf16.h>

// ---------------------------------------------------------------------------
// Hyena sequence mixer for MI455X (gfx1250, wave32, WMMA).
// B=8, L=2048, D=768, ORDER=128, EMB=5, FFT=4096.
// Pipeline:
//   1. filter_kernel : tiny MLP -> k[d][l]             (VALU)
//   2. fftk_kernel   : Kf[d] = fft(k[d], 4096)         (LDS FFT)
//   3. gemm_in       : u @ in_W + b -> x1/x2/v (B,D,L) (WMMA f16->f32)
//   4. convgate      : depthwise-4 causal conv + gate  (elementwise)
//   5. fftconv       : y = ifft(fft(v)*Kf) + v*Dbias, *x2c (LDS FFT)
//   6. gemm_out      : y @ out_W + b -> out (B,L,D)    (WMMA f16->f32)
// GEMM geometry: WG tile 128(M) x 256(N), 8 waves, each wave 64x64
// (16 x v_wmma_f32_16x16x32_f16 per K-step, K-step = 32).
// Staging: batched b128 global loads (12 in flight) -> f16 LDS tiles.
// ---------------------------------------------------------------------------

typedef __attribute__((ext_vector_type(16))) _Float16 v16h;
typedef __attribute__((ext_vector_type(8)))  _Float16 v8h;
typedef __attribute__((ext_vector_type(4)))  _Float16 v4h;
typedef __attribute__((ext_vector_type(8)))  float    v8f;
typedef __attribute__((ext_vector_type(4)))  float    v4f;

#define BATCH 8
#define SEQL  2048
#define DMODEL 768
#define ORDER 128
#define FFT_N 4096
#define FFT_LOG 12
#define PI_F 3.14159265358979323846f
#define MIN_DECAY (-3.0701134573253944f)   /* log(0.01)/1.5 */
#define MAX_DECAY (-15.350567286626972f)   /* log(0.01)/0.3 */

#define LDA 40   /* padded LDS row stride in halves (80B -> conflict-free b128) */

// ------------------------------ WMMA helpers -------------------------------
// A fragment (16x32 f16): lane m = lane&15, lane>>4 selects K-half.
// Per ISA: lane<16 holds K {0..7, 16..23}; lane>=16 holds K {8..15, 24..31}.
__device__ __forceinline__ v16h load_a_frag(const _Float16* As, int row0, int lane) {
  int m  = row0 + (lane & 15);
  int kh = (lane >> 4) << 3;            // 0 or 8
  const _Float16* p = As + m * LDA;
  v8h lo = *(const v8h*)(p + kh);       // K = kh .. kh+7
  v8h hi = *(const v8h*)(p + 16 + kh);  // K = 16+kh .. 16+kh+7
  return __builtin_shufflevector(lo, hi, 0,1,2,3,4,5,6,7,8,9,10,11,12,13,14,15);
}

// B fragment (32x16 f16), LDS holds B transposed [n][k]:
// lane n = lane&15; lane<16 -> K {0..7, 16..23}, lane>=16 -> K {8..15, 24..31}.
__device__ __forceinline__ v16h load_b_frag(const _Float16* Bt, int col0, int lane) {
  int n  = col0 + (lane & 15);
  int k0 = (lane >> 4) << 3;            // 0 or 8
  const _Float16* p = Bt + n * LDA + k0;
  v8h lo = *(const v8h*)(p);
  v8h hi = *(const v8h*)(p + 16);
  return __builtin_shufflevector(lo, hi, 0,1,2,3,4,5,6,7,8,9,10,11,12,13,14,15);
}

#define WMMA_F32_F16(C, A, Bv) \
  (C) = __builtin_amdgcn_wmma_f32_16x16x32_f16(false, (A), false, (Bv), (short)0, (C), false, false)

// ------------------------- 1. filter MLP kernel ----------------------------
__global__ __launch_bounds__(ORDER)
void filter_kernel(const float* __restrict__ z, const float* __restrict__ sin_freq,
                   const float* __restrict__ eo_mat, const float* __restrict__ eo_bias,
                   const float* __restrict__ oo1_mat, const float* __restrict__ oo1_bias,
                   const float* __restrict__ oo2_mat, const float* __restrict__ oo2_bias,
                   const float* __restrict__ oh_mat, float* __restrict__ kbuf) {
  __shared__ float hA[ORDER];
  __shared__ float hB[ORDER];
  const int l = blockIdx.x;
  const int o = threadIdx.x;
  const float f = sin_freq[o];

  float acc = eo_bias[o];
  #pragma unroll
  for (int e = 0; e < 5; ++e) acc += z[l * 5 + e] * eo_mat[e * ORDER + o];
  hA[o] = __sinf(f * acc);
  __syncthreads();

  acc = oo1_bias[o];
  for (int p = 0; p < ORDER; ++p) acc += hA[p] * oo1_mat[p * ORDER + o];
  hB[o] = __sinf(f * acc);
  __syncthreads();

  acc = oo2_bias[o];
  for (int p = 0; p < ORDER; ++p) acc += hB[p] * oo2_mat[p * ORDER + o];
  hA[o] = __sinf(f * acc);
  __syncthreads();

  const float t = (float)l / (float)(SEQL - 1);
  #pragma unroll
  for (int q = 0; q < DMODEL / ORDER; ++q) {
    const int d = o + q * ORDER;
    float k = 0.f;
    for (int p = 0; p < ORDER; ++p) k += hA[p] * oh_mat[p * DMODEL + d];
    const float delta = fabsf(MIN_DECAY + (MAX_DECAY - MIN_DECAY) * ((float)d / (float)(DMODEL - 1)));
    k *= __expf(-t * delta);            // SHIFT == 0
    kbuf[(size_t)d * SEQL + l] = k;     // k stored transposed: (D, L)
  }
}

// ------------------------------ FFT in LDS ---------------------------------
__device__ __forceinline__ void fft4096(float* re, float* im, int tid, int nthr, float sign) {
  for (unsigned i = tid; i < FFT_N; i += nthr) {
    unsigned j = __brev(i) >> (32 - FFT_LOG);
    if (j > i) {
      float tr = re[i]; re[i] = re[j]; re[j] = tr;
      float ti = im[i]; im[i] = im[j]; im[j] = ti;
    }
  }
  __syncthreads();
  for (int s = 1; s <= FFT_LOG; ++s) {
    const int half = 1 << (s - 1);
    const int len  = half << 1;
    const float ab = sign * (-2.0f * PI_F) / (float)len;
    for (int t = tid; t < FFT_N / 2; t += nthr) {
      const int blk = t >> (s - 1);
      const int k   = t & (half - 1);
      const int i0  = blk * len + k;
      const int i1  = i0 + half;
      float wr, wi;
      __sincosf(ab * (float)k, &wi, &wr);
      const float xr = re[i1], xi = im[i1];
      const float tr = xr * wr - xi * wi;
      const float ti = xr * wi + xi * wr;
      const float ur = re[i0], ui = im[i0];
      re[i0] = ur + tr; im[i0] = ui + ti;
      re[i1] = ur - tr; im[i1] = ui - ti;
    }
    __syncthreads();
  }
}

// ------------------------- 2. filter spectrum ------------------------------
__global__ __launch_bounds__(256)
void fftk_kernel(const float* __restrict__ kbuf, float* __restrict__ kf) {
  __shared__ float re[FFT_N];
  __shared__ float im[FFT_N];
  const int d = blockIdx.x;
  const int tid = threadIdx.x;
  for (int i = tid; i < FFT_N; i += 256) {
    re[i] = (i < SEQL) ? kbuf[(size_t)d * SEQL + i] : 0.f;
    im[i] = 0.f;
  }
  __syncthreads();
  fft4096(re, im, tid, 256, +1.f);
  for (int i = tid; i < FFT_N; i += 256) {
    kf[((size_t)d * FFT_N + i) * 2 + 0] = re[i];
    kf[((size_t)d * FFT_N + i) * 2 + 1] = im[i];
  }
}

// ------------------------- 3. input projection GEMM ------------------------
// C(16384 x 2304) = U(16384 x 768) @ W(768 x 2304) + bias; output written
// transposed into x1/x2/v planes laid out (B, D, L).
__global__ __launch_bounds__(256)
void gemm_in_kernel(const float* __restrict__ U, const float* __restrict__ W,
                    const float* __restrict__ bias,
                    float* __restrict__ x1raw, float* __restrict__ x2raw,
                    float* __restrict__ vraw) {
  __shared__ _Float16 As[128 * LDA];    // [m][k]
  __shared__ _Float16 Bt[256 * LDA];    // [n][k] (transposed)
  const int K = DMODEL, N = 3 * DMODEL;
  const int tid  = threadIdx.x;
  const int lane = tid & 31;
  const int wave = tid >> 5;
  const int mBase = blockIdx.x * 128;
  const int nBase = blockIdx.y * 256;
  const int wm = (wave >> 2) * 64;      // 2 wave-rows
  const int wn = (wave & 3) * 64;       // 4 wave-cols

  // Hoisted per-thread staging bases (fixed trip counts: 4 A + 8 B loads).
  // A: element (r, c4), r = (tid>>3) + it*32, c4 = (tid&7)*4.
  const float*    pU   = U + (size_t)(mBase + (tid >> 3)) * K + ((tid & 7) << 2);
  _Float16*       sA   = As + (tid >> 3) * LDA + ((tid & 7) << 2);
  // B: element (n4.., cc), n4 = (tid&63)*4, cc = (tid>>6) + it*4.
  const float*    pW   = W + (size_t)(tid >> 6) * N + nBase + ((tid & 63) << 2);
  _Float16*       sB   = Bt + ((tid & 63) << 2) * LDA + (tid >> 6);

  v8f c[4][4] = {};
  for (int k0 = 0; k0 < K; k0 += 32) {
    // ---- batched global loads (12 x b128 in flight) ----
    v4f fa[4], fb[8];
    #pragma unroll
    for (int it = 0; it < 4; ++it)
      fa[it] = *(const v4f*)(pU + (size_t)it * 32 * K + k0);
    #pragma unroll
    for (int it = 0; it < 8; ++it)
      fb[it] = *(const v4f*)(pW + (size_t)(k0 + it * 4) * N);
    // ---- convert + LDS store ----
    #pragma unroll
    for (int it = 0; it < 4; ++it) {
      v4h h; h[0] = (_Float16)fa[it][0]; h[1] = (_Float16)fa[it][1];
             h[2] = (_Float16)fa[it][2]; h[3] = (_Float16)fa[it][3];
      *(v4h*)(sA + it * 32 * LDA) = h;
    }
    #pragma unroll
    for (int it = 0; it < 8; ++it) {
      _Float16* q = sB + it * 4;
      q[0 * LDA] = (_Float16)fb[it][0];
      q[1 * LDA] = (_Float16)fb[it][1];
      q[2 * LDA] = (_Float16)fb[it][2];
      q[3 * LDA] = (_Float16)fb[it][3];
    }
    if (k0 + 32 < K) {   // prefetch next K-step tiles
      __builtin_prefetch(pU + (size_t)((tid & 3) * 32) * K + k0 + 32, 0, 1);
      __builtin_prefetch(pW + (size_t)(k0 + 32 + (tid & 31)) * N, 0, 1);
    }
    __syncthreads();
    v16h a[4], bf[4];
    #pragma unroll
    for (int t = 0; t < 4; ++t) a[t]  = load_a_frag(As, wm + 16 * t, lane);
    #pragma unroll
    for (int t = 0; t < 4; ++t) bf[t] = load_b_frag(Bt, wn + 16 * t, lane);
    #pragma unroll
    for (int im = 0; im < 4; ++im)
      #pragma unroll
      for (int jn = 0; jn < 4; ++jn)
        WMMA_F32_F16(c[im][jn], a[im], bf[jn]);
    __syncthreads();
  }

  // epilogue: plane/d0 are block constants (256-wide tile never crosses 768)
  const int b  = mBase >> 11;
  const int lB = mBase & (SEQL - 1);
  const int plane = nBase / DMODEL;
  const int d0 = nBase - plane * DMODEL;
  float* __restrict__ dst = (plane == 0) ? x1raw : (plane == 1) ? x2raw : vraw;
  #pragma unroll
  for (int im = 0; im < 4; ++im)
    #pragma unroll
    for (int jn = 0; jn < 4; ++jn) {
      const int d  = d0 + wn + jn * 16 + (lane & 15);
      const int l0 = lB + wm + im * 16 + ((lane >> 4) << 3);
      const float bn = bias[nBase + wn + jn * 16 + (lane & 15)];
      float* p = dst + (((size_t)b * DMODEL + d) << 11) + l0;
      v4f lo = { c[im][jn][0] + bn, c[im][jn][1] + bn, c[im][jn][2] + bn, c[im][jn][3] + bn };
      v4f hi = { c[im][jn][4] + bn, c[im][jn][5] + bn, c[im][jn][6] + bn, c[im][jn][7] + bn };
      *(v4f*)p = lo;            // 8 consecutive L elements per lane
      *(v4f*)(p + 4) = hi;
    }
}

// ------------------------- 4. depthwise conv + gate ------------------------
__global__ __launch_bounds__(256)
void convgate_kernel(const float* __restrict__ x1raw, const float* __restrict__ x2raw,
                     const float* __restrict__ vraw,
                     const float* __restrict__ x1_s, const float* __restrict__ x2_s,
                     const float* __restrict__ v_s,
                     const float* __restrict__ x1_sb, const float* __restrict__ x2_sb,
                     const float* __restrict__ v_sb,
                     float* __restrict__ vg, float* __restrict__ x2c) {
  const size_t idx = (size_t)blockIdx.x * 256 + threadIdx.x;   // over B*D*L
  const int l = (int)(idx & (SEQL - 1));
  const size_t chan = idx >> 11;
  const int d = (int)(chan % DMODEL);
  float a1 = x1_sb[d], a2 = x2_sb[d], av = v_sb[d];
  #pragma unroll
  for (int j = 0; j < 4; ++j) {
    const int t = l + j - 3;                 // causal: pad 3 on the left
    if (t >= 0) {
      const size_t src = (chan << 11) + t;
      a1 += x1raw[src] * x1_s[d * 4 + j];
      a2 += x2raw[src] * x2_s[d * 4 + j];
      av += vraw[src]  * v_s[d * 4 + j];
    }
  }
  vg[idx]  = av * a1;
  x2c[idx] = a2;
}

// ------------------------- 5. FFT convolution ------------------------------
__global__ __launch_bounds__(256)
void fftconv_kernel(const float* __restrict__ vg, const float* __restrict__ x2c,
                    const float* __restrict__ kf, const float* __restrict__ D_bias,
                    float* __restrict__ ybuf) {
  __shared__ float re[FFT_N];
  __shared__ float im[FFT_N];
  const int bd = blockIdx.x;            // b*D + d
  const int d  = bd % DMODEL;
  const int tid = threadIdx.x;
  const float* v = vg + ((size_t)bd << 11);

  for (int i = tid; i < FFT_N; i += 256) {
    re[i] = (i < SEQL) ? v[i] : 0.f;
    im[i] = 0.f;
  }
  __syncthreads();
  fft4096(re, im, tid, 256, +1.f);      // forward
  for (int i = tid; i < FFT_N; i += 256) {
    const float ar = re[i], ai = im[i];
    const float br = kf[((size_t)d * FFT_N + i) * 2 + 0];
    const float bi = kf[((size_t)d * FFT_N + i) * 2 + 1];
    re[i] = ar * br - ai * bi;
    im[i] = ar * bi + ai * br;
  }
  __syncthreads();
  fft4096(re, im, tid, 256, -1.f);      // inverse (scale below)
  const float inv = 1.0f / (float)FFT_N;
  const float db  = D_bias[d];
  for (int l = tid; l < SEQL; l += 256) {
    float y = re[l] * inv + v[l] * db;
    y *= x2c[((size_t)bd << 11) + l];
    ybuf[((size_t)bd << 11) + l] = y;
  }
}

// ------------------------- 6. output projection GEMM -----------------------
// out(16384 x 768) = Y(16384 x 768) @ W(768 x 768) + bias,
// Y read from (B, D, L) layout (transpose while staging into LDS).
__global__ __launch_bounds__(256)
void gemm_out_kernel(const float* __restrict__ ybuf, const float* __restrict__ W,
                     const float* __restrict__ bias, float* __restrict__ out) {
  __shared__ _Float16 As[128 * LDA];
  __shared__ _Float16 Bt[256 * LDA];
  const int K = DMODEL, N = DMODEL;
  const int tid  = threadIdx.x;
  const int lane = tid & 31;
  const int wave = tid >> 5;
  const int mBase = blockIdx.x * 128;
  const int nBase = blockIdx.y * 256;
  const int b = mBase >> 11;
  const int lBase = mBase & (SEQL - 1);
  const int wm = (wave >> 2) * 64;
  const int wn = (wave & 3) * 64;

  // A: element (r4.., cc), r4 = (tid&31)*4 (along L), cc = (tid>>5) + it*8.
  const float* pY = ybuf + (((size_t)b * DMODEL + (tid >> 5)) << 11) + lBase + ((tid & 31) << 2);
  _Float16*    sA = As + ((tid & 31) << 2) * LDA + (tid >> 5);
  // B: same pattern as gemm_in.
  const float* pW = W + (size_t)(tid >> 6) * N + nBase + ((tid & 63) << 2);
  _Float16*    sB = Bt + ((tid & 63) << 2) * LDA + (tid >> 6);

  v8f c[4][4] = {};
  for (int k0 = 0; k0 < K; k0 += 32) {
    v4f fa[4], fb[8];
    #pragma unroll
    for (int it = 0; it < 4; ++it)
      fa[it] = *(const v4f*)(pY + ((size_t)(k0 + it * 8) << 11));
    #pragma unroll
    for (int it = 0; it < 8; ++it)
      fb[it] = *(const v4f*)(pW + (size_t)(k0 + it * 4) * N);
    #pragma unroll
    for (int it = 0; it < 4; ++it) {
      _Float16* q = sA + it * 8;
      q[0 * LDA] = (_Float16)fa[it][0];
      q[1 * LDA] = (_Float16)fa[it][1];
      q[2 * LDA] = (_Float16)fa[it][2];
      q[3 * LDA] = (_Float16)fa[it][3];
    }
    #pragma unroll
    for (int it = 0; it < 8; ++it) {
      _Float16* q = sB + it * 4;
      q[0 * LDA] = (_Float16)fb[it][0];
      q[1 * LDA] = (_Float16)fb[it][1];
      q[2 * LDA] = (_Float16)fb[it][2];
      q[3 * LDA] = (_Float16)fb[it][3];
    }
    if (k0 + 32 < K) {
      __builtin_prefetch(pY + ((size_t)(k0 + 32 + (tid & 31)) << 11), 0, 1);
      __builtin_prefetch(pW + (size_t)(k0 + 32 + (tid & 31)) * N, 0, 1);
    }
    __syncthreads();
    v16h a[4], bf[4];
    #pragma unroll
    for (int t = 0; t < 4; ++t) a[t]  = load_a_frag(As, wm + 16 * t, lane);
    #pragma unroll
    for (int t = 0; t < 4; ++t) bf[t] = load_b_frag(Bt, wn + 16 * t, lane);
    #pragma unroll
    for (int im = 0; im < 4; ++im)
      #pragma unroll
      for (int jn = 0; jn < 4; ++jn)
        WMMA_F32_F16(c[im][jn], a[im], bf[jn]);
    __syncthreads();
  }
  #pragma unroll
  for (int im = 0; im < 4; ++im)
    #pragma unroll
    for (int jn = 0; jn < 4; ++jn)
      #pragma unroll
      for (int r = 0; r < 8; ++r) {
        const int m = mBase + wm + im * 16 + ((lane >> 4) << 3) + r;
        const int n = nBase + wn + jn * 16 + (lane & 15);
        out[(size_t)m * N + n] = c[im][jn][r] + bias[n];   // lane-contiguous in n
      }
}

// ------------------------------- launch ------------------------------------
extern "C" void kernel_launch(void* const* d_in, const int* in_sizes, int n_in,
                              void* d_out, int out_size, void* d_ws, size_t ws_size,
                              hipStream_t stream) {
  const float* u        = (const float*)d_in[0];
  const float* in_W     = (const float*)d_in[1];
  const float* in_b     = (const float*)d_in[2];
  const float* out_W    = (const float*)d_in[3];
  const float* out_b    = (const float*)d_in[4];
  const float* x1_s     = (const float*)d_in[5];
  const float* x2_s     = (const float*)d_in[6];
  const float* v_s      = (const float*)d_in[7];
  const float* x1_sb    = (const float*)d_in[8];
  const float* x2_sb    = (const float*)d_in[9];
  const float* v_sb     = (const float*)d_in[10];
  const float* D_bias   = (const float*)d_in[11];
  const float* z        = (const float*)d_in[12];
  const float* sin_freq = (const float*)d_in[13];
  const float* eo_mat   = (const float*)d_in[14];
  const float* eo_bias  = (const float*)d_in[15];
  const float* oo1_mat  = (const float*)d_in[16];
  const float* oo1_bias = (const float*)d_in[17];
  const float* oo2_mat  = (const float*)d_in[18];
  const float* oo2_bias = (const float*)d_in[19];
  const float* oh_mat   = (const float*)d_in[20];

  const size_t plane = (size_t)BATCH * DMODEL * SEQL;   // 12,582,912 floats
  float* ws    = (float*)d_ws;
  float* x1raw = ws;
  float* x2raw = x1raw + plane;
  float* vraw  = x2raw + plane;
  float* vg    = vraw  + plane;
  float* x2c   = vg    + plane;
  float* ybuf  = x2c   + plane;
  float* kbuf  = ybuf  + plane;                          // D*L
  float* kf    = kbuf  + (size_t)DMODEL * SEQL;          // D*FFT_N*2

  // 1. filter + 2. its spectrum
  filter_kernel<<<SEQL, ORDER, 0, stream>>>(z, sin_freq, eo_mat, eo_bias,
                                            oo1_mat, oo1_bias, oo2_mat, oo2_bias,
                                            oh_mat, kbuf);
  fftk_kernel<<<DMODEL, 256, 0, stream>>>(kbuf, kf);

  // 3. input projection (WMMA), writes transposed planes
  dim3 g1((BATCH * SEQL) / 128, (3 * DMODEL) / 256);
  gemm_in_kernel<<<g1, 256, 0, stream>>>(u, in_W, in_b, x1raw, x2raw, vraw);

  // 4. depthwise conv + gate
  convgate_kernel<<<(unsigned)(plane / 256), 256, 0, stream>>>(
      x1raw, x2raw, vraw, x1_s, x2_s, v_s, x1_sb, x2_sb, v_sb, vg, x2c);

  // 5. FFT long convolution + D skip + x2 gate
  fftconv_kernel<<<BATCH * DMODEL, 256, 0, stream>>>(vg, x2c, kf, D_bias, ybuf);

  // 6. output projection (WMMA)
  dim3 g2((BATCH * SEQL) / 128, DMODEL / 256);
  gemm_out_kernel<<<g2, 256, 0, stream>>>(ybuf, out_W, out_b, (float*)d_out);
}
